// LSTMencoder_24953759990006
// MI455X (gfx1250) — matrix operand order repaired and checked by reading, not compiled, not run on hardware
//
#include <hip/hip_runtime.h>
#include <hip/hip_bf16.h>

// CDNA5 (gfx1250) wave32 WMMA + helper vector types
typedef __attribute__((ext_vector_type(16))) _Float16 v16h;
typedef __attribute__((ext_vector_type(8)))  float    v8f;
typedef __attribute__((ext_vector_type(2)))  _Float16 h2v;
typedef __attribute__((ext_vector_type(4)))  _Float16 h4v;
typedef __attribute__((ext_vector_type(2)))  float    f2v;
typedef __attribute__((ext_vector_type(4)))  float    f4v;

#define TDIM 2048
#define BDIM 256
#define HDIM 32
#define G4H  128   // 4*H gate count

// ---- Fast activations (CDNA5 v_tanh_f32 transcendental) ----
#if __has_builtin(__builtin_amdgcn_tanhf)
__device__ __forceinline__ float tanh_fast(float x) { return __builtin_amdgcn_tanhf(x); }
#elif __has_builtin(__builtin_amdgcn_tanh_f32)
__device__ __forceinline__ float tanh_fast(float x) { return __builtin_amdgcn_tanh_f32(x); }
#else
__device__ __forceinline__ float tanh_fast(float x) {
    float e = __expf(2.0f * x);
    return 1.0f - 2.0f * __builtin_amdgcn_rcpf(e + 1.0f);
}
#endif
__device__ __forceinline__ float sig_fast(float x) {
    return 0.5f + 0.5f * tanh_fast(0.5f * x);               // division-free sigmoid
}

// Inverse of ISA 7.12.2 "16-bit A-Matrix 16x32" layout: (m,k) -> (lane, elem)
__device__ __forceinline__ void frag_slot(int m, int k, int& ln, int& e) {
    int b  = (k >= 16) ? 1 : 0;
    int r  = k & 15;
    int hi = r >> 3;
    int q  = r & 7;
    int v  = 4 * b + (q >> 1);
    e  = 2 * v + (q & 1);
    ln = m + 16 * hi;
}

// Bidirectional LSTM layer scan. Grid: (B/16)*2 blocks; blockIdx&1 = direction.
// Block: 256 threads = 8 wave32s; wave w owns gate columns [16w, 16w+16).
__global__ void __launch_bounds__(256)
bilstm_scan(const float* __restrict__ outseq,
            const float* __restrict__ inseq,
            const float* __restrict__ xh,
            const float* __restrict__ W_ih,   // [2][128][F]
            const float* __restrict__ W_hh,   // [2][128][32]
            const float* __restrict__ b_ih,   // [2][128]
            const float* __restrict__ b_hh,   // [2][128]
            float* __restrict__ dst,
            int layer)
{
    // x double-buffered + h, staged in LDS already in A-fragment order.
    __shared__ __attribute__((aligned(32))) _Float16 xF[2][32][32]; // 4 KB
    __shared__ __attribute__((aligned(32))) _Float16 hF[32][16];    // 1 KB
    __shared__ float gates[16][128];                                // 8 KB

    const int tid  = threadIdx.x;
    const int lane = tid & 31;
    const int hi   = lane >> 4;
    const int nl   = lane & 15;
    const int n    = (tid >> 5) * 16 + nl;  // global gate index 0..127
    const int dir  = blockIdx.x & 1;
    const int b0   = (blockIdx.x >> 1) * 16;
    const int F    = (layer == 0) ? 32 : 64;
    const int tsign = dir ? -1 : 1;
    const int t0    = dir ? (TDIM - 1) : 0;

    const float* Wi = W_ih + (size_t)dir * G4H * F;
    const float* Wh = W_hh + (size_t)dir * G4H * HDIM;
    const float  bias = b_ih[dir * G4H + n] + b_hh[dir * G4H + n];

    // ---- Resident weight B fragments: elem e <-> K = 8*hi + e (+8 if e>=8), N = lane&15 ----
    union { v16h v; _Float16 h[16]; } bh, bx0, bx1;
    #pragma unroll
    for (int e = 0; e < 16; ++e) {
        int k = 8 * hi + e + ((e >= 8) ? 8 : 0);
        bh.h[e]  = (_Float16)Wh[(size_t)n * HDIM + k];
        bx0.h[e] = (_Float16)Wi[(size_t)n * F + k];
        bx1.h[e] = (layer != 0) ? (_Float16)Wi[(size_t)n * F + 32 + k] : (_Float16)0.0f;
    }

    // ---- Strength-reduced per-thread pointers/slots ----
    // x prefetch source (layer0: float2/thread; layer1: float4/thread)
    const float* xptr;
    ptrdiff_t    xdelta;   // floats per step
    int sm, sf;
    if (layer == 0) {
        int idx = tid * 2; sm = idx >> 5; sf = idx & 31;
        int b = b0 + sm;
        if (sf < 16) {
            xptr   = outseq + ((size_t)b * TDIM + t0) * 16 + sf;
            xdelta = (ptrdiff_t)tsign * 16;
        } else {
            xptr   = inseq + ((size_t)b * TDIM + (TDIM - 1 - t0)) * 16 + (sf - 16);
            xdelta = -(ptrdiff_t)tsign * 16;
        }
    } else {
        int idx = tid * 4; sm = idx >> 6; sf = idx & 63;
        xptr   = xh + ((size_t)t0 * BDIM + b0 + sm) * 64 + sf;
        xdelta = (ptrdiff_t)tsign * BDIM * 64;
    }
    int sln, se; frag_slot(sm, sf & 31, sln, se);
    const int sslot = ((sf >> 5) * 16) + se;     // staging elem within xF row

    // output pointer (each thread owns pair (om, oj), (om, oj+1))
    const int om = (tid * 2) >> 5, oj = (tid * 2) & 31;
    float*    optr;
    ptrdiff_t odelta;
    if (layer == 0) {
        optr   = dst + ((size_t)t0 * BDIM + b0 + om) * 64 + dir * 32 + oj;  // h1 [T][B][64]
        odelta = (ptrdiff_t)tsign * BDIM * 64;
    } else {
        optr   = dst + ((size_t)(b0 + om) * TDIM + t0) * 64 + dir * 32 + oj; // out [B][T][64]
        odelta = (ptrdiff_t)tsign * 64;
    }
    int oln, oe; frag_slot(om, oj, oln, oe);     // hF slot for this thread's pair

    // ---- Init: h fragment zero in LDS, cell state zero in REGISTERS ----
    for (int idx = tid; idx < 512; idx += 256)
        hF[idx >> 4][idx & 15] = (_Float16)0.0f;
    float c0 = 0.0f, c1 = 0.0f;

    // ---- Prologue: prefetch x(t0) into registers ----
    float xr0, xr1, xr2 = 0.0f, xr3 = 0.0f;
    if (layer == 0) {
        f2v a = *(const f2v*)xptr; xr0 = a.x; xr1 = a.y;
    } else {
        f4v a = *(const f4v*)xptr; xr0 = a.x; xr1 = a.y; xr2 = a.z; xr3 = a.w;
    }
    xptr += xdelta;

    int cur = 0;
    for (int step = 0; step < TDIM; ++step) {
        // ---- Stage prefetched registers into xF[cur] (one LDS store/thread) ----
        if (layer == 0) {
            h2v hx; hx.x = (_Float16)xr0; hx.y = (_Float16)xr1;
            *(h2v*)&xF[cur][sln][sslot] = hx;
        } else {
            h4v hx; hx.x = (_Float16)xr0; hx.y = (_Float16)xr1;
                    hx.z = (_Float16)xr2; hx.w = (_Float16)xr3;
            *(h4v*)&xF[cur][sln][sslot] = hx;
        }
        __syncthreads();   // barrier A: xF[cur] + hF(step) visible

        // ---- Software pipeline: issue x(t+1) global loads (latency hidden) ----
        if (step + 1 < TDIM) {
            if (layer == 0) {
                f2v a = *(const f2v*)xptr; xr0 = a.x; xr1 = a.y;
            } else {
                f4v a = *(const f4v*)xptr; xr0 = a.x; xr1 = a.y; xr2 = a.z; xr3 = a.w;
            }
            xptr += xdelta;
        }

        // ---- A fragments (contiguous 32B per lane) + WMMA chain ----
        v16h ax0 = *(const v16h*)&xF[cur][lane][0];
        v16h ah  = *(const v16h*)&hF[lane][0];

        union { v8f v; float f[8]; } acc;
        #pragma unroll
        for (int r = 0; r < 8; ++r) acc.f[r] = bias;

        acc.v = __builtin_amdgcn_wmma_f32_16x16x32_f16(
            false, ax0, false, bx0.v, (short)0, acc.v, false, false);
        if (layer != 0) {
            v16h ax1 = *(const v16h*)&xF[cur][lane][16];
            acc.v = __builtin_amdgcn_wmma_f32_16x16x32_f16(
                false, ax1, false, bx1.v, (short)0, acc.v, false, false);
        }
        acc.v = __builtin_amdgcn_wmma_f32_16x16x32_f16(
            false, ah, false, bh.v, (short)0, acc.v, false, false);

        // ---- Scatter C fragment: row m = r + 8*hi, col n ----
        #pragma unroll
        for (int r = 0; r < 8; ++r)
            gates[r + 8 * hi][n] = acc.f[r];
        __syncthreads();   // barrier B: gates ready

        // ---- Elementwise cell update for pair (om, oj), (om, oj+1) ----
        f2v gi = *(const f2v*)&gates[om][oj];
        f2v gf = *(const f2v*)&gates[om][32 + oj];
        f2v gg = *(const f2v*)&gates[om][64 + oj];
        f2v go = *(const f2v*)&gates[om][96 + oj];
        c0 = sig_fast(gf.x) * c0 + sig_fast(gi.x) * tanh_fast(gg.x);
        c1 = sig_fast(gf.y) * c1 + sig_fast(gi.y) * tanh_fast(gg.y);
        float h0 = sig_fast(go.x) * tanh_fast(c0);
        float h1 = sig_fast(go.y) * tanh_fast(c1);

        h2v hh; hh.x = (_Float16)h0; hh.y = (_Float16)h1;
        *(h2v*)&hF[oln][oe] = hh;                 // recurrent h for next step

        f2v ho; ho.x = h0; ho.y = h1;
        if (layer == 0) {
            *(f2v*)optr = ho;                     // h1 stays L2-resident for layer 1
        } else {
            __builtin_nontemporal_store(ho, (f2v*)optr);  // final output: stream past L2
        }
        optr += odelta;

        cur ^= 1;
        // barrier A of the next iteration orders hF writes vs. fragment reads
    }
}

extern "C" void kernel_launch(void* const* d_in, const int* in_sizes, int n_in,
                              void* d_out, int out_size, void* d_ws, size_t ws_size,
                              hipStream_t stream) {
    (void)in_sizes; (void)n_in; (void)out_size; (void)ws_size;

    const float* outseq = (const float*)d_in[0];  // [B,T,16]
    const float* inseq  = (const float*)d_in[1];  // [B,T,16]
    const float* W_ih0  = (const float*)d_in[2];  // [2,128,32]
    const float* W_hh0  = (const float*)d_in[3];  // [2,128,32]
    const float* b_ih0  = (const float*)d_in[4];  // [2,128]
    const float* b_hh0  = (const float*)d_in[5];  // [2,128]
    const float* W_ih1  = (const float*)d_in[6];  // [2,128,64]
    const float* W_hh1  = (const float*)d_in[7];  // [2,128,32]
    const float* b_ih1  = (const float*)d_in[8];  // [2,128]
    const float* b_hh1  = (const float*)d_in[9];  // [2,128]

    float* out = (float*)d_out;                   // [B,T,64]
    float* h1  = (float*)d_ws;                    // [T,B,64] = 128 MB scratch

    dim3 grid((BDIM / 16) * 2);                   // 32 workgroups
    dim3 block(256);                              // 8 wave32s

    bilstm_scan<<<grid, block, 0, stream>>>(outseq, inseq, nullptr,
                                            W_ih0, W_hh0, b_ih0, b_hh0, h1, 0);
    bilstm_scan<<<grid, block, 0, stream>>>(nullptr, nullptr, h1,
                                            W_ih1, W_hh1, b_ih1, b_hh1, out, 1);
}
